// self_attention_82059645157894
// MI455X (gfx1250) — matrix operand order repaired and checked by reading, not compiled, not run on hardware
//
#include <hip/hip_runtime.h>

// ---------------------------------------------------------------------------
// scores = (x_q @ Wq^T) @ (x_k @ Wk^T)^T / sqrt(1024)
// B=8, L=2048, D=E=1024.  All GEMMs run on v_wmma_f32_16x16x32_bf16 (fp32 acc).
// fp32 inputs converted to bf16 via native v_cvt_pk_bf16_f32 during LDS staging;
// Q/K intermediates kept bf16 in workspace (2 x 32 MB).  LDS double-buffered:
// one barrier per K-step, global loads for tile t+1 overlap WMMA of tile t.
// ---------------------------------------------------------------------------

typedef __attribute__((ext_vector_type(4)))  __bf16 v4bf;
typedef __attribute__((ext_vector_type(8)))  __bf16 v8bf;
typedef __attribute__((ext_vector_type(16))) __bf16 v16bf;
typedef __attribute__((ext_vector_type(8)))  float  v8f;

#define BM 128          // block tile M
#define BN 128          // block tile N
#define BK 32           // K per LDS stage == WMMA K
#define LDS_PITCH 40    // 32 bf16 + 8 pad (80B row stride -> 20-bank skew)

__device__ __forceinline__ v16bf cat16(v8bf lo, v8bf hi) {
  return __builtin_shufflevector(lo, hi, 0,1,2,3,4,5,6,7,8,9,10,11,12,13,14,15);
}

// A fragment, 16x32 bf16 (ISA 7.12.2): lane<16 -> row M=lane, K {0..7, 16..23};
// lane>=16 -> row M=lane-16, K {8..15, 24..31}.
__device__ __forceinline__ v16bf frag_a(const __bf16* t, int m0, int lane) {
  const __bf16* row = t + (unsigned)(m0 + (lane & 15)) * LDS_PITCH;
  const int half = (lane >> 4) << 3;                 // 0 or 8
  v8bf lo = *(const v8bf*)(row + half);
  v8bf hi = *(const v8bf*)(row + 16 + half);
  return cat16(lo, hi);
}

// B fragment, 32x16 bf16: lane holds column N=lane&15, K = (lane<16 ? {0..7, 16..23} : {8..15, 24..31}).
// B[k][n] = Wtile[n][k] (row-major [N,K] tile) -> each lane reads two runs of 8 contiguous K.
__device__ __forceinline__ v16bf frag_b(const __bf16* t, int n0, int lane) {
  const __bf16* row = t + (unsigned)(n0 + (lane & 15)) * LDS_PITCH;
  const int half = (lane >> 4) << 3;                 // 0 or 8
  v8bf lo = *(const v8bf*)(row + half);
  v8bf hi = *(const v8bf*)(row + 16 + half);
  return cat16(lo, hi);
}

// ---------------------------------------------------------------------------
// Kernel 1: Y[M,1024](bf16) = X[M,1024](f32) @ W[1024,1024](f32)^T
// ---------------------------------------------------------------------------
__global__ __launch_bounds__(256)
void proj_bf16_kernel(const float* __restrict__ X,
                      const float* __restrict__ W,
                      __bf16* __restrict__ Y) {
  __shared__ __align__(16) __bf16 At[2][BM * LDS_PITCH];
  __shared__ __align__(16) __bf16 Bt[2][BN * LDS_PITCH];

  const int tid   = threadIdx.x;
  const int lane  = tid & 31;
  const int wave  = tid >> 5;
  const int waveM = wave & 3;          // 4 M-groups x 32 rows
  const int waveN = wave >> 2;         // 2 N-groups x 64 cols
  const int m_blk = blockIdx.x * BM;
  const int n_blk = blockIdx.y * BN;

  const float* Xp = X + (size_t)m_blk * 1024;
  const float* Wp = W + (size_t)n_blk * 1024;

  // staging coordinates for this thread (4 float4 per matrix per tile)
  float4 ra[4], rb[4];
  auto gload = [&](int kt) {
#pragma unroll
    for (int i = 0; i < 4; ++i) {
      int idx = tid + i * 256;              // float4 index in 128x32 tile
      int r = idx >> 3;                     // 8 float4 per row
      int c = (idx & 7) << 2;               // col (floats)
      ra[i] = *(const float4*)(Xp + (size_t)r * 1024 + kt + c);
      rb[i] = *(const float4*)(Wp + (size_t)r * 1024 + kt + c);
    }
  };
  auto lstore = [&](int buf) {
#pragma unroll
    for (int i = 0; i < 4; ++i) {
      int idx = tid + i * 256;
      int r = idx >> 3;
      int c = (idx & 7) << 2;
      v4bf ha, hb;
      ha[0] = (__bf16)ra[i].x; ha[1] = (__bf16)ra[i].y;
      ha[2] = (__bf16)ra[i].z; ha[3] = (__bf16)ra[i].w;
      hb[0] = (__bf16)rb[i].x; hb[1] = (__bf16)rb[i].y;
      hb[2] = (__bf16)rb[i].z; hb[3] = (__bf16)rb[i].w;
      *(v4bf*)(At[buf] + (unsigned)r * LDS_PITCH + c) = ha;
      *(v4bf*)(Bt[buf] + (unsigned)r * LDS_PITCH + c) = hb;
    }
  };

  v8f acc[2][4];
#pragma unroll
  for (int mi = 0; mi < 2; ++mi)
#pragma unroll
    for (int ni = 0; ni < 4; ++ni)
#pragma unroll
      for (int r = 0; r < 8; ++r) acc[mi][ni][r] = 0.0f;

  // prologue: stage tile 0
  gload(0);
  lstore(0);
  __syncthreads();

  int buf = 0;
  for (int kt = 0; kt < 1024; kt += BK) {
    const bool has_next = (kt + BK) < 1024;
    if (has_next) gload(kt + BK);           // overlap with WMMA below

    v16bf a0 = frag_a(At[buf], waveM * 32,      lane);
    v16bf a1 = frag_a(At[buf], waveM * 32 + 16, lane);
#pragma unroll
    for (int ni = 0; ni < 4; ++ni) {
      v16bf b = frag_b(Bt[buf], waveN * 64 + ni * 16, lane);
      acc[0][ni] = __builtin_amdgcn_wmma_f32_16x16x32_bf16(
          false, a0, false, b, (short)0, acc[0][ni], false, false);
      acc[1][ni] = __builtin_amdgcn_wmma_f32_16x16x32_bf16(
          false, a1, false, b, (short)0, acc[1][ni], false, false);
    }

    if (has_next) {
      lstore(buf ^ 1);                      // write the other buffer
      __syncthreads();                      // single barrier per K-step
      buf ^= 1;
    }
  }

  // ---- epilogue: C VGPR r -> M = m0 + r + 8*(lane>=16), N = n0 + (lane&15)
#pragma unroll
  for (int mi = 0; mi < 2; ++mi)
#pragma unroll
    for (int ni = 0; ni < 4; ++ni) {
      int m0 = m_blk + waveM * 32 + mi * 16 + ((lane >> 4) << 3);
      int n  = n_blk + waveN * 64 + ni * 16 + (lane & 15);
#pragma unroll
      for (int r = 0; r < 8; ++r)
        Y[(size_t)(m0 + r) * 1024 + n] = (__bf16)acc[mi][ni][r];
    }
}

// ---------------------------------------------------------------------------
// Kernel 2: S[b,L,L](f32) = scale * Q[b,L,E](bf16) @ K[b,L,E](bf16)^T
// ---------------------------------------------------------------------------
__global__ __launch_bounds__(256)
void scores_bf16_kernel(const __bf16* __restrict__ Q,
                        const __bf16* __restrict__ Km,
                        float* __restrict__ S,
                        int L, int E, float scale) {
  __shared__ __align__(16) __bf16 At[2][BM * LDS_PITCH];
  __shared__ __align__(16) __bf16 Bt[2][BN * LDS_PITCH];

  const int tid   = threadIdx.x;
  const int lane  = tid & 31;
  const int wave  = tid >> 5;
  const int waveM = wave & 3;
  const int waveN = wave >> 2;
  const int m_blk = blockIdx.x * BM;
  const int n_blk = blockIdx.y * BN;

  const size_t batch_off = (size_t)blockIdx.z * (size_t)L * (size_t)E;
  const __bf16* Qp = Q  + batch_off + (size_t)m_blk * E;
  const __bf16* Kp = Km + batch_off + (size_t)n_blk * E;
  float* Sp = S + (size_t)blockIdx.z * (size_t)L * (size_t)L;

  uint4 rq[2], rk[2];
  auto gload = [&](int et) {
#pragma unroll
    for (int i = 0; i < 2; ++i) {
      int idx = tid + i * 256;              // 16B-chunk index in 128x32 tile
      int r = idx >> 2;                     // 4 chunks per row
      int c = (idx & 3) << 3;               // col (bf16 elems)
      rq[i] = *(const uint4*)(Qp + (size_t)r * E + et + c);
      rk[i] = *(const uint4*)(Kp + (size_t)r * E + et + c);
    }
  };
  auto lstore = [&](int buf) {
#pragma unroll
    for (int i = 0; i < 2; ++i) {
      int idx = tid + i * 256;
      int r = idx >> 2;
      int c = (idx & 3) << 3;
      *(uint4*)(At[buf] + (unsigned)r * LDS_PITCH + c) = rq[i];
      *(uint4*)(Bt[buf] + (unsigned)r * LDS_PITCH + c) = rk[i];
    }
  };

  v8f acc[2][4];
#pragma unroll
  for (int mi = 0; mi < 2; ++mi)
#pragma unroll
    for (int ni = 0; ni < 4; ++ni)
#pragma unroll
      for (int r = 0; r < 8; ++r) acc[mi][ni][r] = 0.0f;

  gload(0);
  lstore(0);
  __syncthreads();

  int buf = 0;
  for (int et = 0; et < E; et += BK) {
    const bool has_next = (et + BK) < E;
    if (has_next) gload(et + BK);

    v16bf a0 = frag_a(At[buf], waveM * 32,      lane);
    v16bf a1 = frag_a(At[buf], waveM * 32 + 16, lane);
#pragma unroll
    for (int ni = 0; ni < 4; ++ni) {
      v16bf b = frag_b(Bt[buf], waveN * 64 + ni * 16, lane);
      acc[0][ni] = __builtin_amdgcn_wmma_f32_16x16x32_bf16(
          false, a0, false, b, (short)0, acc[0][ni], false, false);
      acc[1][ni] = __builtin_amdgcn_wmma_f32_16x16x32_bf16(
          false, a1, false, b, (short)0, acc[1][ni], false, false);
    }

    if (has_next) {
      lstore(buf ^ 1);
      __syncthreads();
      buf ^= 1;
    }
  }

#pragma unroll
  for (int mi = 0; mi < 2; ++mi)
#pragma unroll
    for (int ni = 0; ni < 4; ++ni) {
      int m0 = m_blk + waveM * 32 + mi * 16 + ((lane >> 4) << 3);
      int n  = n_blk + waveN * 64 + ni * 16 + (lane & 15);
#pragma unroll
      for (int r = 0; r < 8; ++r)
        Sp[(size_t)(m0 + r) * L + n] = acc[mi][ni][r] * scale;
    }
}

// ---------------------------------------------------------------------------
extern "C" void kernel_launch(void* const* d_in, const int* in_sizes, int n_in,
                              void* d_out, int out_size, void* d_ws, size_t ws_size,
                              hipStream_t stream) {
  const float* x_q = (const float*)d_in[0];   // [8, 2048, 1024]
  const float* x_k = (const float*)d_in[1];   // [8, 2048, 1024]
  const float* Wq  = (const float*)d_in[2];   // [1024, 1024]
  const float* Wk  = (const float*)d_in[3];   // [1024, 1024]
  float* out = (float*)d_out;                 // [8, 2048, 2048]

  const int B = 8, L = 2048, E = 1024;
  const int M = B * L;                        // 16384 rows for projections

  __bf16* Qb = (__bf16*)d_ws;                 // 32 MB bf16
  __bf16* Kb = Qb + (size_t)M * E;            // 32 MB bf16

  dim3 blk(256);
  dim3 g1(M / BM, E / BN);                    // (128, 8)
  proj_bf16_kernel<<<g1, blk, 0, stream>>>(x_q, Wq, Qb);
  proj_bf16_kernel<<<g1, blk, 0, stream>>>(x_k, Wk, Kb);

  dim3 g2(L / BM, L / BN, B);                 // (16, 16, 8)
  scores_bf16_kernel<<<g2, blk, 0, stream>>>(Qb, Kb, out, L, E,
                                             0.03125f /* 1/sqrt(1024) */);
}
